// RoleAwareGraphTransformer_23158463660604
// MI455X (gfx1250) — matrix operand order repaired and checked by reading, not compiled, not run on hardware
//
#include <hip/hip_runtime.h>
#include <hip/hip_bf16.h>
#include <cmath>

// ---------------------------------------------------------------------------
// RoleAwareGraphTransformer forward on gfx1250 (CDNA5, wave32, WMMA)
//
// Dims: N=20000, E=320000 per relation, R=5, IN=64, PE=32, TPE=16, IN0=112
//       HID=256, HEADS=4, HEAD_DIM=64.
// rel_agg() in the reference receives R identical copies of conv_sum ->
// softmax is uniform -> output == conv_sum exactly, so it is skipped
// (aggW/aggb/agga unused by construction).
// ---------------------------------------------------------------------------

#define R_REL 5
#define IN_DIM 64
#define PE_DIM 32
#define IN0 112
#define KP0 128          // IN0 padded to a multiple of 32 for WMMA
#define HID 256
#define HEADS 4
#define HDIM 64
#define NEG_SLOPE 0.2f

typedef __attribute__((ext_vector_type(16))) _Float16 v16h;
typedef __attribute__((ext_vector_type(8)))  float    v8f;

union AFrag { v16h v; _Float16 h[16]; };
union BFrag { v16h v; _Float16 h[16]; uint4 q[2]; };
union CFrag { v8f  v; float    f[8];  };

// ---------------- utility kernels ----------------

__global__ void k_fill(float* __restrict__ p, long n, float v) {
    long i = (long)blockIdx.x * blockDim.x + threadIdx.x;
    if (i < n) p[i] = v;
}

__global__ void k_relu(float* __restrict__ p, long n) {
    long i = (long)blockIdx.x * blockDim.x + threadIdx.x;
    if (i < n) p[i] = fmaxf(p[i], 0.0f);
}

// conv[n][c] = sum_r bias[r][c]   (HeteroConv 'sum' of per-relation biases)
__global__ void k_conv_bias_init(const float* __restrict__ b, float* __restrict__ conv, long total) {
    long i = (long)blockIdx.x * blockDim.x + threadIdx.x;
    if (i >= total) return;
    int c = (int)(i % HID);
    float s = 0.0f;
#pragma unroll
    for (int r = 0; r < R_REL; ++r) s += b[r * HID + c];
    conv[i] = s;
}

// ---------------- pearl PE: mean-neighbor aggregation ----------------

__global__ void k_pearl_scatter(const float* __restrict__ x, const int* __restrict__ EI,
                                float* __restrict__ agg, float* __restrict__ deg,
                                long totalRE, int E) {
    long gid = (long)blockIdx.x * blockDim.x + threadIdx.x;
    long slot = gid >> 6;           // edge index across all relations
    int  c    = (int)(gid & 63);
    if (slot >= totalRE) return;
    int r = (int)(slot / E);
    int e = (int)(slot % E);
    const int* EIr = EI + (size_t)r * 2 * E;
    int src = EIr[e];
    int dst = EIr[E + e];
    atomicAdd(&agg[(size_t)dst * IN_DIM + c], x[(size_t)src * IN_DIM + c]);
    if (c == 0) atomicAdd(&deg[dst], 1.0f);
}

// h0[n] = concat(x[n], tanh(mean_agg @ pearl_W + pearl_b), time_pe(date[n]), zeros(16))
__global__ void k_build_h0(const float* __restrict__ x, const int* __restrict__ date,
                           const float* __restrict__ agg, const float* __restrict__ deg,
                           const float* __restrict__ pW, const float* __restrict__ pb,
                           float* __restrict__ h0, int N) {
    int nid = blockIdx.x;
    int t   = threadIdx.x;
    if (nid >= N) return;
    __shared__ float s_an[IN_DIM];
    float dg = fmaxf(deg[nid], 1.0f);
    if (t < IN_DIM) {
        s_an[t] = agg[(size_t)nid * IN_DIM + t] / dg;
        h0[(size_t)nid * KP0 + t] = x[(size_t)nid * IN_DIM + t];
    }
    __syncthreads();
    if (t >= 64 && t < 96) {
        int j = t - 64;                    // PE output 0..31
        float s = pb[j];
#pragma unroll 8
        for (int c = 0; c < IN_DIM; ++c) s += s_an[c] * pW[c * PE_DIM + j];
        h0[(size_t)nid * KP0 + 64 + j] = tanhf(s);
    } else if (t >= 96 && t < 112) {
        int j  = t - 96;                   // 0..15 : 8 sin then 8 cos
        int jj = j & 7;
        float tt  = (float)date[nid] / 10000.0f;
        float div = __expf((float)(2 * jj) * (-logf(10000.0f) / 16.0f));
        float arg = tt * div;
        h0[(size_t)nid * KP0 + t] = (j < 8) ? sinf(arg) : cosf(arg);
    } else if (t >= 112) {
        h0[(size_t)nid * KP0 + t] = 0.0f;  // zero pad cols 112..127
    }
}

// ---------------- WMMA GEMM ----------------
// C[M,Nout] = A[M,Kp] * W[Kact,Nout] (+bias)(+relu), f16 WMMA, f32 accumulate.
// Block = 256 threads = 8 waves. Block tile = 128 rows x 32 cols.
// Weight slab staged in LDS, swizzled so each lane's B fragment is 32
// contiguous bytes (2 x ds_load_b128). Each wave: 2 WMMAs / K-step, shared A.

__global__ void k_wmma_gemm(const float* __restrict__ A, const float* __restrict__ W,
                            const float* __restrict__ bias, float* __restrict__ C,
                            int M, int Kp, int Kact, int Nout, int fuse_relu) {
    __shared__ _Float16 Bs[2 * 256 * 16];        // two 16-col slabs, <=16KB
    const int tid  = threadIdx.x;
    const int lane = tid & 31;
    const int wave = tid >> 5;
    const int col_base = blockIdx.y * 32;
    const int row_base = blockIdx.x * 128 + wave * 16;
    const int nsteps   = Kp >> 5;
    const int slabSz   = Kp * 16;

    // stage swizzled: slab t, step s, K-half hi, col n -> 16 contiguous halves (j=0..15)
    // linear LDS offset: t*slabSz + s*512 + hi*256 + n*16 + j
    for (int i = tid; i < Kp * 32; i += 256) {
        int t   = i / slabSz;
        int rem = i - t * slabSz;
        int k = rem >> 4, n = rem & 15;
        int s = k >> 5, kk = k & 31;
        int hi = (kk >> 3) & 1, j = (kk & 7) + ((kk >> 4) << 3);
        float v = (k < Kact) ? W[(size_t)k * Nout + col_base + t * 16 + n] : 0.0f;
        Bs[t * slabSz + s * 512 + hi * 256 + n * 16 + j] = (_Float16)v;
    }
    __syncthreads();

    const int m  = lane & 15;
    const int hi = lane >> 4;
    const int rowA = row_base + m;
    const bool rok = rowA < M;
    // Out-of-range rows read row 0 (valid, finite). A row m feeds only D row m,
    // and D rows >= M are never stored, so no zero-fill / predication is needed.
    const float* Arow = A + (size_t)(rok ? rowA : 0) * Kp;
    const int kbA   = hi * 8;            // A: lanes 0-15 K{0-7,16-23}; 16-31 K{8-15,24-31}
    const int laneB = hi * 256 + m * 16; // B fragment base within a step block

    CFrag acc0 = {}, acc1 = {};

    for (int s = 0; s < nsteps; ++s) {
        const int ks = s << 5;
        const float4* Ap0 = (const float4*)(Arow + ks + kbA);
        const float4* Ap1 = (const float4*)(Arow + ks + 16 + kbA);
        float4 f0 = Ap0[0], f1 = Ap0[1];
        float4 g0 = Ap1[0], g1 = Ap1[1];

        AFrag a;
        a.h[0]  = (_Float16)f0.x; a.h[1]  = (_Float16)f0.y;
        a.h[2]  = (_Float16)f0.z; a.h[3]  = (_Float16)f0.w;
        a.h[4]  = (_Float16)f1.x; a.h[5]  = (_Float16)f1.y;
        a.h[6]  = (_Float16)f1.z; a.h[7]  = (_Float16)f1.w;
        a.h[8]  = (_Float16)g0.x; a.h[9]  = (_Float16)g0.y;
        a.h[10] = (_Float16)g0.z; a.h[11] = (_Float16)g0.w;
        a.h[12] = (_Float16)g1.x; a.h[13] = (_Float16)g1.y;
        a.h[14] = (_Float16)g1.z; a.h[15] = (_Float16)g1.w;

        BFrag b0, b1;
        const uint4* bp0 = (const uint4*)(Bs + s * 512 + laneB);
        const uint4* bp1 = (const uint4*)(Bs + slabSz + s * 512 + laneB);
        b0.q[0] = bp0[0]; b0.q[1] = bp0[1];
        b1.q[0] = bp1[0]; b1.q[1] = bp1[1];

        if (ks + 32 < Kp) __builtin_prefetch(Arow + ks + 32, 0, 1);  // global_prefetch_b8

        acc0.v = __builtin_amdgcn_wmma_f32_16x16x32_f16(
            false, a.v, false, b0.v, (short)0, acc0.v, false, false);
        acc1.v = __builtin_amdgcn_wmma_f32_16x16x32_f16(
            false, a.v, false, b1.v, (short)0, acc1.v, false, false);
    }

    // store both 16-col tiles; C/D layout: lane half selects M offset 0/8
#pragma unroll
    for (int t = 0; t < 2; ++t) {
        const CFrag& acc = t ? acc1 : acc0;
        const int col = col_base + t * 16 + m;
        const float ba = bias ? bias[col] : 0.0f;
#pragma unroll
        for (int i = 0; i < 8; ++i) {
            int row = row_base + hi * 8 + i;
            if (row < M) {
                float v = acc.f[i] + ba;
                if (fuse_relu) v = fmaxf(v, 0.0f);
                C[(size_t)row * Nout + col] = v;
            }
        }
    }
}

// ---------------- GATv2 edge kernels (one relation at a time) ----------------

__device__ __forceinline__ void atomicMaxF(float* addr, float val) {
    if (val >= 0.0f) atomicMax((int*)addr, __float_as_int(val));
    else             atomicMin((unsigned int*)addr, __float_as_uint(val));
}

__global__ void k_edge_score(const float* __restrict__ xl, const float* __restrict__ xr,
                             const float* __restrict__ att_r, const int* __restrict__ EIr,
                             float* __restrict__ esc, float* __restrict__ emax, int E) {
    long idx = (long)blockIdx.x * blockDim.x + threadIdx.x;
    if (idx >= (long)E * HEADS) return;
    int e = (int)(idx >> 2);
    int h = (int)(idx & 3);
    int src = EIr[e];
    int dst = EIr[E + e];
    const float* pl = xl + (size_t)src * HID + h * HDIM;
    const float* pr = xr + (size_t)dst * HID + h * HDIM;
    const float* pa = att_r + h * HDIM;
    float s = 0.0f;
#pragma unroll 8
    for (int c = 0; c < HDIM; ++c) {
        float mm = pl[c] + pr[c];
        mm = (mm > 0.0f) ? mm : NEG_SLOPE * mm;
        s += mm * pa[c];
    }
    esc[idx] = s;
    atomicMaxF(&emax[(size_t)dst * HEADS + h], s);
}

__global__ void k_edge_expsum(float* __restrict__ esc, const float* __restrict__ emax,
                              float* __restrict__ den, const int* __restrict__ EIr, int E) {
    long idx = (long)blockIdx.x * blockDim.x + threadIdx.x;
    if (idx >= (long)E * HEADS) return;
    int e = (int)(idx >> 2);
    int h = (int)(idx & 3);
    int dst = EIr[E + e];
    float mx = emax[(size_t)dst * HEADS + h];
    if (mx < -3.0e37f) mx = 0.0f;        // jnp.where(isfinite(emax), emax, 0)
    float ex = __expf(esc[idx] - mx);
    esc[idx] = ex;
    atomicAdd(&den[(size_t)dst * HEADS + h], ex);
}

__global__ void k_edge_aggregate(const float* __restrict__ xl, const float* __restrict__ esc,
                                 const float* __restrict__ den, const int* __restrict__ EIr,
                                 float* __restrict__ conv, int E) {
    int e = blockIdx.x;
    int c = threadIdx.x;
    int src = EIr[e];
    int dst = EIr[E + e];
    int h = c >> 6;
    float alpha = esc[(size_t)e * HEADS + h] / (den[(size_t)dst * HEADS + h] + 1e-16f);
    atomicAdd(&conv[(size_t)dst * HID + c], xl[(size_t)src * HID + c] * alpha);
}

// ---------------- final head: out = hmid @ W2 + b2, [N,128] -> [N,3] ----------------

__global__ void k_out_head(const float* __restrict__ hmid, const float* __restrict__ W2,
                           const float* __restrict__ b2, float* __restrict__ out, int N) {
    long i = (long)blockIdx.x * blockDim.x + threadIdx.x;
    if (i >= (long)N * 3) return;
    int nrow = (int)(i / 3);
    int j    = (int)(i % 3);
    float s = b2[j];
#pragma unroll 8
    for (int c = 0; c < 128; ++c) s += hmid[(size_t)nrow * 128 + c] * W2[c * 3 + j];
    out[i] = s;
}

// ---------------- host orchestration ----------------

static inline unsigned cdiv(long a, long b) { return (unsigned)((a + b - 1) / b); }

extern "C" void kernel_launch(void* const* d_in, const int* in_sizes, int n_in,
                              void* d_out, int out_size, void* d_ws, size_t ws_size,
                              hipStream_t stream) {
    const float* x      = (const float*)d_in[0];
    const int*   date   = (const int*)  d_in[1];
    const int*   EI     = (const int*)  d_in[2];   // [R,2,E]
    const float* pW     = (const float*)d_in[3];
    const float* pb     = (const float*)d_in[4];
    const float* Wl0    = (const float*)d_in[5];   // [R,112,256]
    const float* Wr0    = (const float*)d_in[6];
    const float* att0   = (const float*)d_in[7];   // [R,4,64]
    const float* b0     = (const float*)d_in[8];   // [R,256]
    const float* Wl1    = (const float*)d_in[9];   // [R,256,256]
    const float* Wr1    = (const float*)d_in[10];
    const float* att1   = (const float*)d_in[11];
    const float* b1     = (const float*)d_in[12];
    // d_in[13..15] = aggW/aggb/agga : rel_agg is identity on broadcast input -> unused
    const float* oW1    = (const float*)d_in[16];  // [256,128]
    const float* ob1    = (const float*)d_in[17];
    const float* oW2    = (const float*)d_in[18];  // [128,3]
    const float* ob2    = (const float*)d_in[19];
    (void)n_in; (void)out_size; (void)ws_size;

    const int N = in_sizes[0] / IN_DIM;
    const int E = in_sizes[2] / (R_REL * 2);

    // workspace layout
    float* ws   = (float*)d_ws;
    float* h0   = ws;                 ws += (size_t)N * KP0;
    float* agg  = ws;                 ws += (size_t)N * IN_DIM;
    float* deg  = ws;                 ws += (size_t)N;
    float* xl   = ws;                 ws += (size_t)N * HID;
    float* xr   = ws;                 ws += (size_t)N * HID;
    float* h1   = ws;                 ws += (size_t)N * HID;   // layer0 conv sum -> relu
    float* h2   = ws;                 ws += (size_t)N * HID;   // layer1 conv sum -> relu
    float* hmid = ws;                 ws += (size_t)N * 128;
    float* esc  = ws;                 ws += (size_t)E * HEADS;
    float* emax = ws;                 ws += (size_t)N * HEADS;
    float* den  = ws;                 ws += (size_t)N * HEADS;

    const dim3 blk256(256);
    auto fill = [&](float* p, long n, float v) {
        k_fill<<<cdiv(n, 256), blk256, 0, stream>>>(p, n, v);
    };

    // ---- pearl PE + input feature build ----
    fill(agg, (long)N * IN_DIM, 0.0f);
    fill(deg, (long)N, 0.0f);
    k_pearl_scatter<<<cdiv((long)R_REL * E * IN_DIM, 256), blk256, 0, stream>>>(
        x, EI, agg, deg, (long)R_REL * E, E);
    k_build_h0<<<N, 128, 0, stream>>>(x, date, agg, deg, pW, pb, h0, N);

    // ---- two GNN layers ----
    for (int layer = 0; layer < 2; ++layer) {
        const float* hin  = layer == 0 ? h0 : h1;
        float*       conv = layer == 0 ? h1 : h2;
        const int    Kp   = layer == 0 ? KP0 : HID;
        const int    Kact = layer == 0 ? IN0 : HID;
        const float* Wl   = layer == 0 ? Wl0 : Wl1;
        const float* Wr   = layer == 0 ? Wr0 : Wr1;
        const float* attb = layer == 0 ? att0 : att1;
        const float* bb   = layer == 0 ? b0 : b1;

        k_conv_bias_init<<<cdiv((long)N * HID, 256), blk256, 0, stream>>>(bb, conv, (long)N * HID);

        const dim3 ggrid(cdiv(N, 128), HID / 32);
        for (int r = 0; r < R_REL; ++r) {
            const size_t woff = (size_t)r * Kact * HID;
            k_wmma_gemm<<<ggrid, blk256, 0, stream>>>(hin, Wl + woff, nullptr, xl, N, Kp, Kact, HID, 0);
            k_wmma_gemm<<<ggrid, blk256, 0, stream>>>(hin, Wr + woff, nullptr, xr, N, Kp, Kact, HID, 0);

            fill(emax, (long)N * HEADS, -INFINITY);
            fill(den,  (long)N * HEADS, 0.0f);

            const int* EIr = EI + (size_t)r * 2 * E;
            const float* att_r = attb + (size_t)r * HEADS * HDIM;
            k_edge_score<<<cdiv((long)E * HEADS, 256), blk256, 0, stream>>>(xl, xr, att_r, EIr, esc, emax, E);
            k_edge_expsum<<<cdiv((long)E * HEADS, 256), blk256, 0, stream>>>(esc, emax, den, EIr, E);
            k_edge_aggregate<<<E, blk256, 0, stream>>>(xl, esc, den, EIr, conv, E);
        }
        k_relu<<<cdiv((long)N * HID, 256), blk256, 0, stream>>>(conv, (long)N * HID);
        // rel_agg(broadcast(conv)) == conv exactly -> skip
    }

    // ---- output MLP ----
    const dim3 mgrid(cdiv(N, 128), 128 / 32);
    k_wmma_gemm<<<mgrid, blk256, 0, stream>>>(h2, oW1, ob1, hmid, N, HID, HID, 128, 1);
    k_out_head<<<cdiv((long)N * 3, 256), blk256, 0, stream>>>(hmid, oW2, ob2, (float*)d_out, N);
}
